// FastCLIPLoss_6038724018696
// MI455X (gfx1250) — matrix operand order repaired and checked by reading, not compiled, not run on hardware
//
#include <hip/hip_runtime.h>

#define EPSF 1e-14f
#define RHOF 8.0f

typedef _Float16 v16h __attribute__((ext_vector_type(16)));
typedef _Float16 v8h  __attribute__((ext_vector_type(8)));
typedef _Float16 v4h  __attribute__((ext_vector_type(4)));
typedef float    v8f  __attribute__((ext_vector_type(8)));

union Frag { v16h v; v8h h[2]; };

#define BM 128
#define BN 64
#define PITCH 40   // halves per LDS row: 32 data + 8 pad (80B pitch -> conflict-free b128 reads)

struct __align__(16) Smem {
  _Float16 A1[2][BM * PITCH];
  _Float16 A2[2][BM * PITCH];
  _Float16 B1[2][BN * PITCH];
  _Float16 B2[2][BN * PITCH];
  float diag[BM];
  float wim[BM];
  float wtt[BM];
  float red[8];
};

__device__ __forceinline__ float wred(float v) {
  v += __shfl_xor(v, 16, 32);
  v += __shfl_xor(v, 8, 32);
  v += __shfl_xor(v, 4, 32);
  v += __shfl_xor(v, 2, 32);
  v += __shfl_xor(v, 1, 32);
  return v;
}

// ---------------- f32 -> f16 conversion (vectorized) ----------------
__global__ void k_f32_to_f16(const float* __restrict__ src, _Float16* __restrict__ dst, int n4) {
  int i = blockIdx.x * blockDim.x + threadIdx.x;
  if (i < n4) {
    float4 v = reinterpret_cast<const float4*>(src)[i];
    v4h o = { (_Float16)v.x, (_Float16)v.y, (_Float16)v.z, (_Float16)v.w };
    reinterpret_cast<v4h*>(dst)[i] = o;
  }
}

// ---------------- per-row diag dot + weights + accumulator zero ----------------
__global__ void k_prep(const float* __restrict__ rim, const float* __restrict__ rtt,
                       const float* __restrict__ ruim, const float* __restrict__ rutt,
                       float* __restrict__ diag, float* __restrict__ wim, float* __restrict__ wtt,
                       float* __restrict__ acc, int D4, float rw) {
  __shared__ float sred[8];
  const int r = blockIdx.x, t = threadIdx.x;
  float s = 0.f;
  for (int i = t; i < D4; i += blockDim.x) {
    float4 a = reinterpret_cast<const float4*>(rim)[(size_t)r * D4 + i];
    float4 b = reinterpret_cast<const float4*>(rtt)[(size_t)r * D4 + i];
    s += a.x * b.x + a.y * b.y + a.z * b.z + a.w * b.w;
  }
  s = wred(s);
  if ((t & 31) == 0) sred[t >> 5] = s;
  __syncthreads();
  if (t == 0) {
    float tot = 0.f;
#pragma unroll
    for (int w = 0; w < 8; ++w) tot += sred[w];
    diag[r] = tot;
    wim[r] = 1.0f / ((ruim[r] + EPSF) * rw);
    wtt[r] = 1.0f / ((rutt[r] + EPSF) * rw);
    if (r == 0) acc[0] = 0.f;   // re-zeroed each launch (graph-replay safe)
  }
}

// ---------------- async global -> LDS tile staging (no VGPR data regs) ----------------
__device__ __forceinline__ unsigned lds_off32(const void* p) {
  // AMDGPU generic pointers to LDS carry the LDS byte offset in the low 32 bits
  return (unsigned)(uintptr_t)p;
}

__device__ __forceinline__ void async_b128(unsigned ldsoff, unsigned goff, const _Float16* base) {
  asm volatile("global_load_async_to_lds_b128 %0, %1, %2"
               :
               : "v"(ldsoff), "v"(goff), "s"((unsigned long long)(uintptr_t)base)
               : "memory");
}

__device__ __forceinline__ void issue_tiles(Smem& sm, int buf, int tid,
                                            const _Float16* gA1, const _Float16* gA2,
                                            const _Float16* gB1, const _Float16* gB2,
                                            int m0, int n0, int D, int kk) {
  const int k0 = kk * 32;  // half offset along K
#pragma unroll
  for (int i = 0; i < 2; ++i) {
    int idx = tid + i * 256;
    int row = idx >> 2, seg = idx & 3;
    unsigned goff = (unsigned)(((m0 + row) * D + k0 + seg * 8) * 2);   // bytes
    unsigned lo = (unsigned)((row * PITCH + seg * 8) * 2);
    async_b128(lds_off32(&sm.A1[buf][0]) + lo, goff, gA1);
    async_b128(lds_off32(&sm.A2[buf][0]) + lo, goff, gA2);
  }
  {
    int row = tid >> 2, seg = tid & 3;
    unsigned goff = (unsigned)(((n0 + row) * D + k0 + seg * 8) * 2);
    unsigned lo = (unsigned)((row * PITCH + seg * 8) * 2);
    async_b128(lds_off32(&sm.B1[buf][0]) + lo, goff, gB1);
    async_b128(lds_off32(&sm.B2[buf][0]) + lo, goff, gB2);
  }
}

// ---------------- WMMA compute for one K=32 step ----------------
__device__ __forceinline__ void compute_step(Smem& sm, int buf, int wm, int wn, int lane,
                                             v8f c1[2][2], v8f c2[2][2]) {
  const int lm = lane & 15, lh = lane >> 4;
  // A-fragment (16x32 f16): lanes 0-15 hold K 0..7 & 16..23; lanes 16-31 hold K 8..15 & 24..31
  // B-fragment (32x16 f16): lanes 0-15 hold K 0..7 & 16..23; lanes 16-31 hold K 8..15 & 24..31 (column = lane%16)
  Frag a[2], b[2];
  // ---- GEMM 1: rim x tt^T ----
#pragma unroll
  for (int ti = 0; ti < 2; ++ti) {
    int row = wm * 32 + ti * 16 + lm;
    int base = row * PITCH + lh * 8;
    a[ti].h[0] = *reinterpret_cast<const v8h*>(&sm.A1[buf][base]);
    a[ti].h[1] = *reinterpret_cast<const v8h*>(&sm.A1[buf][base + 16]);
  }
#pragma unroll
  for (int tj = 0; tj < 2; ++tj) {
    int col = wn * 32 + tj * 16 + lm;
    int base = col * PITCH + lh * 8;
    b[tj].h[0] = *reinterpret_cast<const v8h*>(&sm.B1[buf][base]);
    b[tj].h[1] = *reinterpret_cast<const v8h*>(&sm.B1[buf][base + 16]);
  }
#pragma unroll
  for (int ti = 0; ti < 2; ++ti)
#pragma unroll
    for (int tj = 0; tj < 2; ++tj)
      c1[ti][tj] = __builtin_amdgcn_wmma_f32_16x16x32_f16(
          false, a[ti].v, false, b[tj].v, (short)0, c1[ti][tj], false, false);
  // ---- GEMM 2: rtt x im^T ----
#pragma unroll
  for (int ti = 0; ti < 2; ++ti) {
    int row = wm * 32 + ti * 16 + lm;
    int base = row * PITCH + lh * 8;
    a[ti].h[0] = *reinterpret_cast<const v8h*>(&sm.A2[buf][base]);
    a[ti].h[1] = *reinterpret_cast<const v8h*>(&sm.A2[buf][base + 16]);
  }
#pragma unroll
  for (int tj = 0; tj < 2; ++tj) {
    int col = wn * 32 + tj * 16 + lm;
    int base = col * PITCH + lh * 8;
    b[tj].h[0] = *reinterpret_cast<const v8h*>(&sm.B2[buf][base]);
    b[tj].h[1] = *reinterpret_cast<const v8h*>(&sm.B2[buf][base + 16]);
  }
#pragma unroll
  for (int ti = 0; ti < 2; ++ti)
#pragma unroll
    for (int tj = 0; tj < 2; ++tj)
      c2[ti][tj] = __builtin_amdgcn_wmma_f32_16x16x32_f16(
          false, a[ti].v, false, b[tj].v, (short)0, c2[ti][tj], false, false);
}

// ---------------- fused dual-GEMM + exp epilogue, reduced to one scalar ----------------
__global__ __launch_bounds__(256)
void k_gemm_epilogue(const _Float16* __restrict__ A1g, const _Float16* __restrict__ B1g,
                     const _Float16* __restrict__ A2g, const _Float16* __restrict__ B2g,
                     const float* __restrict__ diag, const float* __restrict__ wim,
                     const float* __restrict__ wtt, float* __restrict__ acc,
                     const float* __restrict__ lsp, const int* __restrict__ offp,
                     int D, int nK) {
  __shared__ Smem sm;
  const int tid  = threadIdx.x;
  const int lane = tid & 31;
  const int wave = tid >> 5;
  const int wm = wave >> 1;  // 0..3 : 32-row strip within 128-row block tile
  const int wn = wave & 1;   // 0..1 : 32-col strip within 64-col block tile
  const int m0 = blockIdx.y * BM;
  const int n0 = blockIdx.x * BN;

  if (tid < BM) {
    sm.diag[tid] = diag[m0 + tid];
    sm.wim[tid]  = wim[m0 + tid];
    sm.wtt[tid]  = wtt[m0 + tid];
  }

  v8f c1[2][2], c2[2][2];
  const v8f vz = {0.f, 0.f, 0.f, 0.f, 0.f, 0.f, 0.f, 0.f};
#pragma unroll
  for (int i = 0; i < 2; ++i)
#pragma unroll
    for (int j = 0; j < 2; ++j) { c1[i][j] = vz; c2[i][j] = vz; }

  // prologue: async-stage tile 0 into buffer 0
  issue_tiles(sm, 0, tid, A1g, A2g, B1g, B2g, m0, n0, D, 0);

  for (int ks = 0; ks < nK; ++ks) {
    const int nxt = ks + 1;
    if (nxt < nK) {
      // stage next tile into the other buffer, then drain only the current tile's
      // 6 async ops (async loads complete in order; 6 = next tile still in flight)
      issue_tiles(sm, nxt & 1, tid, A1g, A2g, B1g, B2g, m0, n0, D, nxt);
      asm volatile("s_wait_asynccnt 0x6" ::: "memory");
    } else {
      asm volatile("s_wait_asynccnt 0x0" ::: "memory");
    }
    __syncthreads();                       // tile (ks&1) complete for all waves
    compute_step(sm, ks & 1, wm, wn, lane, c1, c2);
    __syncthreads();                       // all reads done before buffer is reused
  }

  // Epilogue: exp((sim - diag)*ls) * row-weight, excluding col == row + offset
  const float ls = lsp[0];
  const int off = offp[0];
  const int lm = lane & 15, lh = lane >> 4;
  float lsum = 0.f;
#pragma unroll
  for (int ti = 0; ti < 2; ++ti) {
#pragma unroll
    for (int tj = 0; tj < 2; ++tj) {
      const int cg = n0 + wn * 32 + tj * 16 + lm;
#pragma unroll
      for (int j = 0; j < 8; ++j) {
        const int rl = wm * 32 + ti * 16 + lh * 8 + j;  // C/D layout: M = j + 8*(lane>>4)
        const int rg = m0 + rl;
        const float d = sm.diag[rl];
        const float mask = (cg == rg + off) ? 0.f : 1.f;
        lsum += __expf((c1[ti][tj][j] - d) * ls) * sm.wim[rl] * mask;
        lsum += __expf((c2[ti][tj][j] - d) * ls) * sm.wtt[rl] * mask;
      }
    }
  }
  lsum = wred(lsum);
  if (lane == 0) sm.red[wave] = lsum;
  __syncthreads();
  if (tid == 0) {
    float s = 0.f;
#pragma unroll
    for (int w = 0; w < 8; ++w) s += sm.red[w];
    atomicAdd(acc, s);
  }
}

// ---------------- final scalar: pg1 mean, log mean, tau terms ----------------
__global__ void k_final(const float* __restrict__ l1im, const float* __restrict__ l1tt,
                        const float* __restrict__ uim, const float* __restrict__ utt,
                        const float* __restrict__ lsp, const float* __restrict__ acc,
                        float* __restrict__ out, int B) {
  __shared__ float sp[8], sg[8];
  const int t = threadIdx.x;
  float p = 0.f, g = 0.f;
  for (int r = t; r < B; r += 256) {
    p += l1im[r] / (uim[r] + EPSF) + l1tt[r] / (utt[r] + EPSF);
    g += __logf(uim[r]) + __logf(utt[r]);
  }
  p = wred(p); g = wred(g);
  if ((t & 31) == 0) { sp[t >> 5] = p; sg[t >> 5] = g; }
  __syncthreads();
  if (t == 0) {
    float tp = 0.f, tg = 0.f;
#pragma unroll
    for (int w = 0; w < 8; ++w) { tp += sp[w]; tg += sg[w]; }
    const float ls = lsp[0];
    const float invB = 1.0f / (float)B;
    float loss = (tp * invB + acc[0] * invB) * 0.5f / ls;   // (mean pg1 + mean pg2)/2 / ls
    loss += RHOF / ls;
    loss += (tg * invB) * 0.5f / ls;
    out[0] = loss;
  }
}

extern "C" void kernel_launch(void* const* d_in, const int* in_sizes, int n_in,
                              void* d_out, int out_size, void* d_ws, size_t ws_size,
                              hipStream_t stream) {
  const float* im   = (const float*)d_in[0];
  const float* tt   = (const float*)d_in[1];
  const float* rim  = (const float*)d_in[2];
  const float* rtt  = (const float*)d_in[3];
  const float* ruim = (const float*)d_in[4];
  const float* rutt = (const float*)d_in[5];
  const float* l1im = (const float*)d_in[6];
  const float* l1tt = (const float*)d_in[7];
  const float* uim  = (const float*)d_in[8];
  const float* utt  = (const float*)d_in[9];
  const float* lsp  = (const float*)d_in[10];
  const int*   offp = (const int*)d_in[11];
  float* out = (float*)d_out;

  const int B = in_sizes[4];            // 4096 (remote_u_im is [B,1])
  const int D = in_sizes[0] / B;        // 1024
  const size_t matBytes = (size_t)B * D * sizeof(_Float16);

  char* ws = (char*)d_ws;
  _Float16* A1h = (_Float16*)(ws);                  // rim f16
  _Float16* B1h = (_Float16*)(ws + matBytes);       // tt  f16
  _Float16* A2h = (_Float16*)(ws + 2 * matBytes);   // rtt f16
  _Float16* B2h = (_Float16*)(ws + 3 * matBytes);   // im  f16
  float* diag = (float*)(ws + 4 * matBytes);
  float* wim  = diag + B;
  float* wtt  = wim + B;
  float* acc  = wtt + B;

  const int n4 = B * D / 4;
  const int cvtBlocks = (n4 + 255) / 256;
  k_f32_to_f16<<<cvtBlocks, 256, 0, stream>>>(rim, A1h, n4);
  k_f32_to_f16<<<cvtBlocks, 256, 0, stream>>>(tt,  B1h, n4);
  k_f32_to_f16<<<cvtBlocks, 256, 0, stream>>>(rtt, A2h, n4);
  k_f32_to_f16<<<cvtBlocks, 256, 0, stream>>>(im,  B2h, n4);

  k_prep<<<B, 256, 0, stream>>>(rim, rtt, ruim, rutt, diag, wim, wtt, acc,
                                D / 4, (float)(B - 1));

  dim3 grid(B / BN, B / BM);
  k_gemm_epilogue<<<grid, 256, 0, stream>>>(A1h, B1h, A2h, B2h, diag, wim, wtt, acc,
                                            lsp, offp, D, D / 32);

  k_final<<<1, 256, 0, stream>>>(l1im, l1tt, uim, utt, lsp, acc, out, B);
}
